// SET_DQN_2156073583116
// MI455X (gfx1250) — matrix operand order repaired and checked by reading, not compiled, not run on hardware
//
#include <hip/hip_runtime.h>
#include <hip/hip_bf16.h>

#define N_IMG 8192
#define K_CL  1024
#define E_EP  32
#define A_ACT 16384
#define D_DIM 1024
#define H_DIM 1024

typedef __attribute__((ext_vector_type(16))) __bf16 bf16x16;
typedef __attribute__((ext_vector_type(8)))  float  f32x8;

union FragAB {
    bf16x16 v;
    uint4   q[2];
};

__device__ __forceinline__ unsigned short f2bf(float f) {
    unsigned int u = __float_as_uint(f);
    u += 0x7FFFu + ((u >> 16) & 1u);        // round-to-nearest-even
    return (unsigned short)(u >> 16);
}

// ---------------------------------------------------------------------------
// Unified BF16 WMMA GEMM:  out = act( A @ B + bias ),  Nout fixed at 1024.
// A is composed of up to 3 K-segments of width 1024; each segment has a base
// pointer (bf16, row stride 1024) and an optional row-gather index array.
// Bt is the transposed weight (Nout=1024 rows x Ktot cols, bf16, row-major).
// Block: 256 threads (8 waves). Block tile 128(M) x 128(N). K step 32.
// Each wave -> 32x64 output (2 A-frags x 4 B-frags = 8 WMMA per K-step).
// B tile is staged with GLOBAL_LOAD_ASYNC_TO_LDS_B128 (ASYNCcnt), A tile with
// regular loads (needs OOB zero-fill + row gather).
// ---------------------------------------------------------------------------
__global__ __launch_bounds__(256)
void wmma_gemm_kernel(const unsigned short* __restrict__ a0, const int* __restrict__ i0,
                      const unsigned short* __restrict__ a1, const int* __restrict__ i1,
                      const unsigned short* __restrict__ a2, const int* __restrict__ i2,
                      const unsigned short* __restrict__ Bt,
                      const float* __restrict__ bias,
                      float* __restrict__ outF, unsigned short* __restrict__ outB,
                      int M, int Ktot, int relu)
{
    __shared__ __align__(16) unsigned short Alds[128 * 40];   // 128 rows, stride 40 (pad)
    __shared__ __align__(16) unsigned short Blds[128 * 40];   // 128 n-rows, stride 40

    const int t    = threadIdx.x;
    const int wave = t >> 5;
    const int lane = t & 31;
    const int lh   = lane >> 4;      // half-wave (0/1)
    const int lq   = lane & 15;
    const int wm   = wave >> 1;      // 0..3  -> which 32-row slice
    const int wn   = wave & 1;       // 0..1  -> which 64-col group

    const int rowBase = blockIdx.x * 128;
    const int colBase = blockIdx.y * 128;

    f32x8 acc[2][4] = {};

    for (int k0 = 0; k0 < Ktot; k0 += 32) {
        const int seg = k0 >> 10;                      // 32 | 1024: never straddles
        const unsigned short* sp = (seg == 0) ? a0 : (seg == 1) ? a1 : a2;
        const int*            ix = (seg == 0) ? i0 : (seg == 1) ? i1 : i2;

        // ---- stage A tile (gathered, segmented, OOB -> zeros): 128x32 ----
#pragma unroll
        for (int cc = 0; cc < 2; ++cc) {
            const int c   = t + cc * 256;
            const int ar  = c >> 2;                    // 0..127 row
            const int ac  = (c & 3) << 3;              // 0,8,16,24 col
            const int agr = rowBase + ar;
            uint4 av = {0u, 0u, 0u, 0u};
            if (agr < M) {
                const int r = ix ? ix[agr] : agr;
                av = *(const uint4*)(sp + ((size_t)r << 10) + (k0 & 1023) + ac);
            }
            *(uint4*)(&Alds[ar * 40 + ac]) = av;
        }

        // ---- stage B tile asynchronously: memory -> LDS, no VGPR bounce ----
#pragma unroll
        for (int cc = 0; cc < 2; ++cc) {
            const int c  = t + cc * 256;
            const int br = c >> 2;                     // 0..127
            const int bc = (c & 3) << 3;               // 0,8,16,24
            unsigned long long gsrc =
                (unsigned long long)(Bt + (size_t)(colBase + br) * Ktot + k0 + bc);
            // low 32 bits of a generic shared-pointer == LDS byte offset
            unsigned ldsOff = (unsigned)(size_t)(&Blds[br * 40 + bc]);
            asm volatile("global_load_async_to_lds_b128 %0, %1, off"
                         :: "v"(ldsOff), "v"(gsrc)
                         : "memory");
        }
        asm volatile("s_wait_asynccnt 0x0" ::: "memory");
        __syncthreads();

        // ---- A fragments: 16x32 bf16, documented CDNA5 per-lane layout ----
        FragAB fa[2];
#pragma unroll
        for (int sA = 0; sA < 2; ++sA) {
            const int row = wm * 32 + sA * 16 + lq;
            fa[sA].q[0] = *(const uint4*)(&Alds[row * 40 + lh * 8]);       // k: half*8..+8
            fa[sA].q[1] = *(const uint4*)(&Alds[row * 40 + 16 + lh * 8]);  // k: 16+half*8..+8
        }
#pragma unroll
        for (int tile = 0; tile < 4; ++tile) {
            FragAB fb;
            const int n = wn * 64 + tile * 16 + lq;
            fb.q[0] = *(const uint4*)(&Blds[n * 40 + lh * 8]);             // k: half*8..+8
            fb.q[1] = *(const uint4*)(&Blds[n * 40 + 16 + lh * 8]);        // k: 16+half*8..+8
#pragma unroll
            for (int sA = 0; sA < 2; ++sA) {
                acc[sA][tile] = __builtin_amdgcn_wmma_f32_16x16x32_bf16(
                    false, fa[sA].v, false, fb.v, (short)0, acc[sA][tile], false, false);
            }
        }
        __syncthreads();
    }

    // ---- epilogue: C layout VGPR i -> (m = i + half*8, n = lane%16) ----
#pragma unroll
    for (int sA = 0; sA < 2; ++sA) {
#pragma unroll
        for (int tile = 0; tile < 4; ++tile) {
            const int ncol = colBase + wn * 64 + tile * 16 + lq;
            const float b  = bias ? bias[ncol] : 0.f;
#pragma unroll
            for (int i = 0; i < 8; ++i) {
                const int mrow = rowBase + wm * 32 + sA * 16 + lh * 8 + i;
                if (mrow < M) {
                    float v = acc[sA][tile][i] + b;
                    if (relu) v = fmaxf(v, 0.f);
                    if (outF) outF[(size_t)mrow * 1024 + ncol] = v;
                    if (outB) outB[(size_t)mrow * 1024 + ncol] = f2bf(v);
                }
            }
        }
    }
}

// ---------------------------------------------------------------------------
// Helper kernels
// ---------------------------------------------------------------------------
__global__ void k_onehot_idx(const float* __restrict__ oh, int rows, int cols,
                             int* __restrict__ out) {
    int r = blockIdx.x * blockDim.x + threadIdx.x;
    if (r >= rows) return;
    const float* p = oh + (size_t)r * cols;
    int best = 0; float bv = p[0];
    for (int c = 1; c < cols; ++c) { float v = p[c]; if (v > bv) { bv = v; best = c; } }
    out[r] = best;
}

// dst[idx[r]*1024 + c] += src[r*1024+c] * (scale ? scale[r] : 1)
__global__ void k_scatter_add(const float* __restrict__ src, const int* __restrict__ idx,
                              const float* __restrict__ scale, float* __restrict__ dst,
                              long total) {
    long i = (long)blockIdx.x * blockDim.x + threadIdx.x;
    if (i >= total) return;
    int r = (int)(i >> 10);
    int c = (int)(i & 1023);
    float v = src[i];
    if (scale) v *= scale[r];
    atomicAdd(dst + ((size_t)idx[r] << 10) + c, v);
}

// dst_bf[i] = bf16( src[i] * (rowscale ? rowscale[i>>10] : 1) )
__global__ void k_cvt_bf(const float* __restrict__ src, const float* __restrict__ rowscale,
                         unsigned short* __restrict__ dst, long total) {
    long i = (long)blockIdx.x * blockDim.x + threadIdx.x;
    if (i >= total) return;
    float v = src[i];
    if (rowscale) v *= rowscale[i >> 10];
    dst[i] = f2bf(v);
}

// W[Kt][1024] (f32) -> Wt[1024][Kt] (bf16)
__global__ void k_tcvt_bf(const float* __restrict__ W, unsigned short* __restrict__ Wt,
                          int Kt, long total) {
    long o = (long)blockIdx.x * blockDim.x + threadIdx.x;
    if (o >= total) return;
    int n = (int)(o / Kt);
    int k = (int)(o - (long)n * Kt);
    Wt[o] = f2bf(W[(size_t)k * 1024 + n]);
}

// out[r] = dot( act(X[r,:]), w ) + bias[0]   (one wave32 per row)
__global__ __launch_bounds__(256)
void k_dot_bias(const float* __restrict__ X, const float* __restrict__ w,
                const float* __restrict__ bias, int relu_in,
                float* __restrict__ out, int M) {
    int wave = threadIdx.x >> 5, lane = threadIdx.x & 31;
    int row = blockIdx.x * 8 + wave;
    if (row >= M) return;
    const float* xr = X + (size_t)row * 1024;
    float s = 0.f;
    for (int c = lane; c < 1024; c += 32) {
        float v = xr[c];
        if (relu_in) v = fmaxf(v, 0.f);
        s += v * w[c];
    }
    for (int off = 16; off; off >>= 1) s += __shfl_xor(s, off, 32);
    if (lane == 0) out[row] = s + bias[0];
}

__global__ void k_exp_scatter(const float* __restrict__ g, const int* __restrict__ idx,
                              float* __restrict__ ge, float* __restrict__ denom, int M) {
    int i = blockIdx.x * blockDim.x + threadIdx.x;
    if (i >= M) return;
    float e = expf(g[i]);
    ge[i] = e;
    atomicAdd(denom + idx[i], e);
}

__global__ void k_seg_div(const float* __restrict__ ge, const int* __restrict__ idx,
                          const float* __restrict__ denom, float sign,
                          float* __restrict__ out, int M) {
    int i = blockIdx.x * blockDim.x + threadIdx.x;
    if (i >= M) return;
    out[i] = sign * ge[i] / denom[idx[i]];
}

// q_expand[e*A + a] = (a_idx[a]==e) ? q_table[a] : 0
__global__ void k_expand(const float* __restrict__ qt, const int* __restrict__ a_idx,
                         float* __restrict__ out, long total) {
    long i = (long)blockIdx.x * blockDim.x + threadIdx.x;
    if (i >= total) return;
    int e = (int)(i >> 14);
    int a = (int)(i & 16383);
    out[i] = (a_idx[a] == e) ? qt[a] : 0.f;
}

// ---------------------------------------------------------------------------
extern "C" void kernel_launch(void* const* d_in, const int* in_sizes, int n_in,
                              void* d_out, int out_size, void* d_ws, size_t ws_size,
                              hipStream_t stream) {
    (void)in_sizes; (void)n_in; (void)out_size; (void)ws_size;

    const float* images   = (const float*)d_in[0];
    const int*   select_i = (const int*)  d_in[1];
    const int*   select_j = (const int*)  d_in[2];
    const float* p_mat    = (const float*)d_in[3];
    const float* r_mat    = (const float*)d_in[4];
    const float* c_mat    = (const float*)d_in[5];
    const float* a_mat    = (const float*)d_in[6];
    const float* Wg1  = (const float*)d_in[7];  const float* bg1  = (const float*)d_in[8];
    const float* Wg2  = (const float*)d_in[9];  const float* bg2  = (const float*)d_in[10];
    const float* Wc1  = (const float*)d_in[11]; const float* bc1  = (const float*)d_in[12];
    const float* Wc2  = (const float*)d_in[13]; const float* bc2  = (const float*)d_in[14];
    const float* Wgs1 = (const float*)d_in[15]; const float* bgs1 = (const float*)d_in[16];
    const float* Wgs2 = (const float*)d_in[17]; const float* bgs2 = (const float*)d_in[18];
    const float* Ws1  = (const float*)d_in[19]; const float* bs1  = (const float*)d_in[20];
    const float* Ws2  = (const float*)d_in[21]; const float* bs2  = (const float*)d_in[22];
    const float* Wa1  = (const float*)d_in[23]; const float* ba1  = (const float*)d_in[24];
    const float* Wa2  = (const float*)d_in[25]; const float* ba2  = (const float*)d_in[26];

    char* ws = (char*)d_ws;
    size_t off = 0;
    auto alloc = [&](size_t bytes) -> char* {
        char* p = ws + off;
        off = (off + bytes + 255) & ~(size_t)255;
        return p;
    };

    int*   p_idx = (int*)alloc((size_t)N_IMG * 4);
    int*   c_idx = (int*)alloc((size_t)K_CL * 4);
    int*   a_idx = (int*)alloc((size_t)A_ACT * 4);
    float* means = (float*)alloc((size_t)K_CL * D_DIM * 4);
    unsigned short* means_bf      = (unsigned short*)alloc((size_t)K_CL * D_DIM * 2);
    unsigned short* images_bf     = (unsigned short*)alloc((size_t)N_IMG * D_DIM * 2);
    unsigned short* Wg1t  = (unsigned short*)alloc((size_t)1024 * 2048 * 2);
    unsigned short* Wc1t  = (unsigned short*)alloc((size_t)1024 * 1024 * 2);
    unsigned short* Wc2t  = (unsigned short*)alloc((size_t)1024 * 1024 * 2);
    unsigned short* Wgs1t = (unsigned short*)alloc((size_t)1024 * 1024 * 2);
    unsigned short* Ws1t  = (unsigned short*)alloc((size_t)1024 * 1024 * 2);
    unsigned short* Ws2t  = (unsigned short*)alloc((size_t)1024 * 1024 * 2);
    unsigned short* Wa1t  = (unsigned short*)alloc((size_t)1024 * 3072 * 2);
    float* h1     = (float*)alloc((size_t)N_IMG * H_DIM * 4);
    float* gN     = (float*)alloc((size_t)N_IMG * 4);
    float* geN    = (float*)alloc((size_t)N_IMG * 4);
    float* gateN  = (float*)alloc((size_t)N_IMG * 4);
    float* denomK = (float*)alloc((size_t)K_CL * 4);
    float* cluster_in = (float*)alloc((size_t)K_CL * D_DIM * 4);
    unsigned short* cluster_in_bf  = (unsigned short*)alloc((size_t)K_CL * D_DIM * 2);
    unsigned short* c1_bf          = (unsigned short*)alloc((size_t)K_CL * H_DIM * 2);
    float* cluster_out_f = (float*)alloc((size_t)K_CL * H_DIM * 4);
    unsigned short* cluster_out_bf = (unsigned short*)alloc((size_t)K_CL * H_DIM * 2);
    float* s1_f   = (float*)alloc((size_t)K_CL * H_DIM * 4);
    float* sgK    = (float*)alloc((size_t)K_CL * 4);
    float* sgeK   = (float*)alloc((size_t)K_CL * 4);
    float* sgateK = (float*)alloc((size_t)K_CL * 4);
    float* edenom = (float*)alloc((size_t)E_EP * 4);
    float* state0 = (float*)alloc((size_t)E_EP * H_DIM * 4);
    unsigned short* state0_bf = (unsigned short*)alloc((size_t)E_EP * H_DIM * 2);
    unsigned short* state1_bf = (unsigned short*)alloc((size_t)E_EP * H_DIM * 2);
    unsigned short* state2_bf = (unsigned short*)alloc((size_t)E_EP * H_DIM * 2);
    float* qA     = (float*)alloc((size_t)A_ACT * 4);
    float* qeA    = (float*)alloc((size_t)A_ACT * 4);
    float* adenom = (float*)alloc((size_t)E_EP * 4);

    float* q_table    = (float*)d_out;
    float* q_expand   = q_table + A_ACT;
    float* action_rep = q_expand + (size_t)E_EP * A_ACT;

    // --- indices from one-hot matrices ---
    k_onehot_idx<<<(N_IMG + 255) / 256, 256, 0, stream>>>(p_mat, N_IMG, K_CL, p_idx);
    k_onehot_idx<<<(K_CL + 255) / 256, 256, 0, stream>>>(c_mat, K_CL, E_EP, c_idx);
    k_onehot_idx<<<(A_ACT + 255) / 256, 256, 0, stream>>>(a_mat, A_ACT, E_EP, a_idx);

    // --- zero all accumulators (every launch; deterministic) ---
    hipMemsetAsync(means,      0, (size_t)K_CL * D_DIM * 4, stream);
    hipMemsetAsync(denomK,     0, (size_t)K_CL * 4,         stream);
    hipMemsetAsync(cluster_in, 0, (size_t)K_CL * D_DIM * 4, stream);
    hipMemsetAsync(edenom,     0, (size_t)E_EP * 4,         stream);
    hipMemsetAsync(state0,     0, (size_t)E_EP * H_DIM * 4, stream);
    hipMemsetAsync(adenom,     0, (size_t)E_EP * 4,         stream);

    const long ndTotal = (long)N_IMG * D_DIM;
    const long kdTotal = (long)K_CL * D_DIM;

    // --- cluster means, bf16 conversions ---
    k_scatter_add<<<(int)((ndTotal + 255) / 256), 256, 0, stream>>>(images, p_idx, nullptr, means, ndTotal);
    k_cvt_bf<<<(int)((kdTotal + 255) / 256), 256, 0, stream>>>(means, r_mat, means_bf, kdTotal);
    k_cvt_bf<<<(int)((ndTotal + 255) / 256), 256, 0, stream>>>(images, nullptr, images_bf, ndTotal);

    // --- transpose+convert all GEMM weights ---
    k_tcvt_bf<<<(int)(((long)1024 * 2048 + 255) / 256), 256, 0, stream>>>(Wg1,  Wg1t,  2048, (long)1024 * 2048);
    k_tcvt_bf<<<(int)(((long)1024 * 1024 + 255) / 256), 256, 0, stream>>>(Wc1,  Wc1t,  1024, (long)1024 * 1024);
    k_tcvt_bf<<<(int)(((long)1024 * 1024 + 255) / 256), 256, 0, stream>>>(Wc2,  Wc2t,  1024, (long)1024 * 1024);
    k_tcvt_bf<<<(int)(((long)1024 * 1024 + 255) / 256), 256, 0, stream>>>(Wgs1, Wgs1t, 1024, (long)1024 * 1024);
    k_tcvt_bf<<<(int)(((long)1024 * 1024 + 255) / 256), 256, 0, stream>>>(Ws1,  Ws1t,  1024, (long)1024 * 1024);
    k_tcvt_bf<<<(int)(((long)1024 * 1024 + 255) / 256), 256, 0, stream>>>(Ws2,  Ws2t,  1024, (long)1024 * 1024);
    k_tcvt_bf<<<(int)(((long)1024 * 3072 + 255) / 256), 256, 0, stream>>>(Wa1,  Wa1t,  3072, (long)1024 * 3072);

    // --- gate MLP: h1 = relu([images, means[p_idx]] @ Wg1 + bg1) ---
    {
        dim3 g((N_IMG + 127) / 128, 8);
        wmma_gemm_kernel<<<g, 256, 0, stream>>>(images_bf, nullptr, means_bf, p_idx, nullptr, nullptr,
                                                Wg1t, bg1, h1, nullptr, N_IMG, 2048, 1);
    }
    k_dot_bias<<<(N_IMG + 7) / 8, 256, 0, stream>>>(h1, Wg2, bg2, 0, gN, N_IMG);
    k_exp_scatter<<<(N_IMG + 255) / 256, 256, 0, stream>>>(gN, p_idx, geN, denomK, N_IMG);
    k_seg_div<<<(N_IMG + 255) / 256, 256, 0, stream>>>(geN, p_idx, denomK, 1.f, gateN, N_IMG);

    // --- gated cluster aggregation + cluster MLP ---
    k_scatter_add<<<(int)((ndTotal + 255) / 256), 256, 0, stream>>>(images, p_idx, gateN, cluster_in, ndTotal);
    k_cvt_bf<<<(int)((kdTotal + 255) / 256), 256, 0, stream>>>(cluster_in, nullptr, cluster_in_bf, kdTotal);
    {
        dim3 g((K_CL + 127) / 128, 8);
        wmma_gemm_kernel<<<g, 256, 0, stream>>>(cluster_in_bf, nullptr, nullptr, nullptr, nullptr, nullptr,
                                                Wc1t, bc1, nullptr, c1_bf, K_CL, 1024, 1);
        wmma_gemm_kernel<<<g, 256, 0, stream>>>(c1_bf, nullptr, nullptr, nullptr, nullptr, nullptr,
                                                Wc2t, bc2, cluster_out_f, cluster_out_bf, K_CL, 1024, 0);
        wmma_gemm_kernel<<<g, 256, 0, stream>>>(cluster_out_bf, nullptr, nullptr, nullptr, nullptr, nullptr,
                                                Wgs1t, bgs1, s1_f, nullptr, K_CL, 1024, 1);
    }
    k_dot_bias<<<(K_CL + 7) / 8, 256, 0, stream>>>(s1_f, Wgs2, bgs2, 0, sgK, K_CL);
    k_exp_scatter<<<(K_CL + 255) / 256, 256, 0, stream>>>(sgK, c_idx, sgeK, edenom, K_CL);
    k_seg_div<<<(K_CL + 255) / 256, 256, 0, stream>>>(sgeK, c_idx, edenom, 1.f, sgateK, K_CL);

    // --- episode state ---
    k_scatter_add<<<(int)(((long)K_CL * H_DIM + 255) / 256), 256, 0, stream>>>(
        cluster_out_f, c_idx, sgateK, state0, (long)K_CL * H_DIM);
    k_cvt_bf<<<(int)(((long)E_EP * H_DIM + 255) / 256), 256, 0, stream>>>(
        state0, nullptr, state0_bf, (long)E_EP * H_DIM);
    {
        dim3 g(1, 8);   // M = 32 fits in one 128-row block
        wmma_gemm_kernel<<<g, 256, 0, stream>>>(state0_bf, nullptr, nullptr, nullptr, nullptr, nullptr,
                                                Ws1t, bs1, nullptr, state1_bf, E_EP, 1024, 1);
        wmma_gemm_kernel<<<g, 256, 0, stream>>>(state1_bf, nullptr, nullptr, nullptr, nullptr, nullptr,
                                                Ws2t, bs2, nullptr, state2_bf, E_EP, 1024, 0);
    }

    // --- pairs GEMM (dominant): [A x 3072] @ Wa1, A rows gathered on the fly ---
    {
        dim3 g((A_ACT + 127) / 128, 8);
        wmma_gemm_kernel<<<g, 256, 0, stream>>>(state2_bf, a_idx,
                                                cluster_out_bf, select_i,
                                                cluster_out_bf, select_j,
                                                Wa1t, ba1, action_rep, nullptr, A_ACT, 3072, 0);
    }

    // --- q head + segment softmax over episodes + expand ---
    k_dot_bias<<<(A_ACT + 7) / 8, 256, 0, stream>>>(action_rep, Wa2, ba2, 1, qA, A_ACT);
    k_exp_scatter<<<(A_ACT + 255) / 256, 256, 0, stream>>>(qA, a_idx, qeA, adenom, A_ACT);
    k_seg_div<<<(A_ACT + 255) / 256, 256, 0, stream>>>(qeA, a_idx, adenom, -1.f, q_table, A_ACT);
    k_expand<<<(int)(((long)E_EP * A_ACT + 255) / 256), 256, 0, stream>>>(
        q_table, a_idx, q_expand, (long)E_EP * A_ACT);
}
